// MultiHeadAttention_1941325218036
// MI455X (gfx1250) — matrix operand order repaired and checked by reading, not compiled, not run on hardware
//
#include <hip/hip_runtime.h>
#include <hip/hip_bf16.h>

typedef __attribute__((ext_vector_type(16))) _Float16 v16h;
typedef __attribute__((ext_vector_type(8)))  _Float16 v8h;
typedef __attribute__((ext_vector_type(8)))  float    v8f;

#define SEQV   4096
#define BATCHV 2
#define EMBED  512
#define NHEADS 8
#define HDIM   64
#define BHV    (BATCHV*NHEADS)   // 16
#define ROWS   (SEQV*BATCHV)     // 8192
#define QKVN   (3*EMBED)         // 1536

// v16h built from two 16-byte halves (A-matrix K mapping used here, contiguous:
// element e -> k = half*16 + e, same order as the B loads)
union AV { v16h v; v8h h[2]; };

__device__ __forceinline__ v8f wmma_f16(v16h a, v16h b, v8f c) {
  return __builtin_amdgcn_wmma_f32_16x16x32_f16(
      false, a, false, b, (short)0, c, false, false);
}

// --- CDNA5 async global->LDS copy (ASYNCcnt-tracked, no VGPR round trip) ---
__device__ __forceinline__ void async_copy16(const _Float16* g, _Float16* l) {
  // low 32 bits of a generic LDS pointer are the LDS byte offset (aperture rule)
  unsigned int loff = (unsigned int)(unsigned long long)l;
  asm volatile("global_load_async_to_lds_b128 %0, %1, off"
               :: "v"(loff), "v"(g) : "memory");
}
__device__ __forceinline__ void stage32(const _Float16* g, _Float16* l) {
  async_copy16(g, l);
  async_copy16(g + 8, l + 8);
}
__device__ __forceinline__ void wait_async0() {
#if __has_builtin(__builtin_amdgcn_s_wait_asynccnt)
  __builtin_amdgcn_s_wait_asynccnt(0);
#else
  asm volatile("s_wait_asynccnt 0x0" ::: "memory");
#endif
}

// ---------------------------------------------------------------------------
// Kernel 0: one-shot f32 -> f16 conversion (x, Win, Wout).
// ---------------------------------------------------------------------------
__global__ void __launch_bounds__(256)
cvt_f16_kernel(const float* __restrict__ in, _Float16* __restrict__ out, int n)
{
  int i = blockIdx.x * 256 + threadIdx.x;
  if (i < n) out[i] = (_Float16)in[i];
}

// ---------------------------------------------------------------------------
// Kernel 1: qkv = xh @ Winh^T + bin, packed into f16 Q[bh][s][d] (pre-scaled),
// K[bh][s][d], V^T[bh][d][s].  One wave: 16(M) x 64(N) tile, K=512.
// ---------------------------------------------------------------------------
__global__ void __launch_bounds__(32)
qkv_pack_kernel(const _Float16* __restrict__ xh, const _Float16* __restrict__ Wh,
                const float* __restrict__ bias,
                _Float16* __restrict__ Qh, _Float16* __restrict__ Kh,
                _Float16* __restrict__ Vt)
{
  const int lane = threadIdx.x;
  const int half = lane >> 4;
  const int m16  = lane & 15;
  const int nbase = blockIdx.x * 64;   // over 1536
  const int rbase = blockIdx.y * 16;   // over 8192

  v8f acc[4] = {};
  const _Float16* xrow = xh + (size_t)(rbase + m16) * EMBED;

  for (int kb = 0; kb < EMBED; kb += 32) {
    AV a;
    a.h[0] = *(const v8h*)(xrow + kb + half * 16);
    a.h[1] = *(const v8h*)(xrow + kb + half * 16 + 8);
    v16h bt[4];
    #pragma unroll
    for (int j = 0; j < 4; ++j)
      bt[j] = *(const v16h*)(Wh + (size_t)(nbase + j * 16 + m16) * EMBED + kb + half * 16);
    #pragma unroll
    for (int j = 0; j < 4; ++j)
      acc[j] = wmma_f16(a.v, bt[j], acc[j]);
  }

  #pragma unroll
  for (int j = 0; j < 4; ++j) {
    const int n   = nbase + j * 16 + m16;
    const float bv = bias[n];
    const int t   = n / EMBED;      // 0=q 1=k 2=v
    const int rem = n % EMBED;
    const int h   = rem / HDIM;
    const int d   = rem % HDIM;
    #pragma unroll
    for (int i = 0; i < 8; ++i) {
      const int r  = rbase + i + half * 8;
      const int s  = r / BATCHV;
      const int bb = r % BATCHV;
      const int bh = bb * NHEADS + h;
      const float v = acc[j][i] + bv;
      if (t == 0)      Qh[((size_t)bh * SEQV + s) * HDIM + d] = (_Float16)(v * 0.125f);
      else if (t == 1) Kh[((size_t)bh * SEQV + s) * HDIM + d] = (_Float16)v;
      else             Vt[((size_t)bh * HDIM + d) * SEQV + s] = (_Float16)v;
    }
  }
}

// ---------------------------------------------------------------------------
// Kernel 2: flash attention, 4 waves / 128 threads per block, 64 q-rows.
// K/V tiles double-buffered in LDS via GLOBAL_LOAD_ASYNC_TO_LDS_B128;
// the async copies for tile n+1 overlap the 8 WMMAs + softmax of tile n.
// ---------------------------------------------------------------------------
__global__ void __launch_bounds__(128)
flash_attn_kernel(const _Float16* __restrict__ Qh, const _Float16* __restrict__ Kh,
                  const _Float16* __restrict__ Vt, _Float16* __restrict__ AO)
{
  __shared__ _Float16 KtS[2][32][64];   // [buf][key][d]      8 KB
  __shared__ _Float16 VlS[2][64][32];   // [buf][d][key]      8 KB
  __shared__ _Float16 Pb[4][16][32];    // per-wave P tiles   4 KB

  const int tid  = threadIdx.x;
  const int wid  = tid >> 5;
  const int lane = tid & 31;
  const int half = lane >> 4;
  const int m16  = lane & 15;
  const int bh    = blockIdx.x;              // 16
  const int qbase = blockIdx.y * 64 + wid * 16;

  // staging assignment: thread copies 32B of K and 32B of V per tile
  const int koff  = tid * 16;                // K tile is one contiguous 4 KB block
  const int vrow  = tid >> 1;
  const int vhalf = tid & 1;
  const _Float16* kbase = Kh + (size_t)bh * SEQV * HDIM;
  const _Float16* vbase = Vt + ((size_t)bh * HDIM + vrow) * SEQV;

  // Q in registers (A-layout), d-slices [0,32) and [32,64)
  const _Float16* qrow = Qh + ((size_t)bh * SEQV + qbase + m16) * HDIM;
  AV qa0, qa1;
  qa0.h[0] = *(const v8h*)(qrow + half * 16);
  qa0.h[1] = *(const v8h*)(qrow + half * 16 + 8);
  qa1.h[0] = *(const v8h*)(qrow + 32 + half * 16);
  qa1.h[1] = *(const v8h*)(qrow + 40 + half * 16);

  v8f o[4] = {};
  float rmax[8], rsum[8];
  #pragma unroll
  for (int i = 0; i < 8; ++i) { rmax[i] = -1e30f; rsum[i] = 0.0f; }

  // prologue: stage tile 0 asynchronously
  stage32(kbase + koff,            &KtS[0][0][0] + koff);
  stage32(vbase + vhalf * 16,      &VlS[0][vrow][vhalf * 16]);
  wait_async0();
  __syncthreads();

  int cur = 0;
  for (int kt = 0; kt < SEQV; kt += 32) {
    const int ktn = kt + 32;
    const bool pre = (ktn < SEQV);
    if (pre) {
      // fire async copies for the next tile into the other buffer
      stage32(kbase + (size_t)ktn * HDIM + koff, &KtS[cur ^ 1][0][0] + koff);
      stage32(vbase + ktn + vhalf * 16,          &VlS[cur ^ 1][vrow][vhalf * 16]);
      if (ktn + 32 < SEQV) {
        __builtin_prefetch(kbase + (size_t)(ktn + 32) * HDIM + koff, 0, 0);
        __builtin_prefetch(vbase + (ktn + 32) + vhalf * 16, 0, 0);
      }
    }

    // --- scores: 16x32 tile = two 16x16 WMMA accumulators
    v8f s0 = {}, s1 = {};
    {
      v16h b00 = *(const v16h*)(&KtS[cur][m16][half * 16]);
      v16h b01 = *(const v16h*)(&KtS[cur][m16][32 + half * 16]);
      v16h b10 = *(const v16h*)(&KtS[cur][16 + m16][half * 16]);
      v16h b11 = *(const v16h*)(&KtS[cur][16 + m16][32 + half * 16]);
      s0 = wmma_f16(qa0.v, b00, s0); s0 = wmma_f16(qa1.v, b01, s0);
      s1 = wmma_f16(qa0.v, b10, s1); s1 = wmma_f16(qa1.v, b11, s1);
    }

    // --- online softmax (row stats replicated across 16 lanes of a half)
    float p0[8], p1[8];
    #pragma unroll
    for (int i = 0; i < 8; ++i) {
      float tm = fmaxf(s0[i], s1[i]);
      #pragma unroll
      for (int m = 8; m >= 1; m >>= 1) tm = fmaxf(tm, __shfl_xor(tm, m, 32));
      const float nm    = fmaxf(rmax[i], tm);
      const float scale = __expf(rmax[i] - nm);
      const float e0 = __expf(s0[i] - nm);
      const float e1 = __expf(s1[i] - nm);
      float ps = e0 + e1;
      #pragma unroll
      for (int m = 8; m >= 1; m >>= 1) ps += __shfl_xor(ps, m, 32);
      rsum[i] = rsum[i] * scale + ps;
      rmax[i] = nm;
      p0[i] = e0; p1[i] = e1;
      #pragma unroll
      for (int j = 0; j < 4; ++j) o[j][i] *= scale;
    }

    // --- C-layout -> A-layout transpose of P through per-wave LDS slice
    #pragma unroll
    for (int i = 0; i < 8; ++i) {
      Pb[wid][i + half * 8][m16]      = (_Float16)p0[i];
      Pb[wid][i + half * 8][16 + m16] = (_Float16)p1[i];
    }
    AV pa;  // wave-private slice: in-order LDS within a wave
    pa.h[0] = *(const v8h*)(&Pb[wid][m16][half * 16]);
    pa.h[1] = *(const v8h*)(&Pb[wid][m16][half * 16 + 8]);

    // --- P @ V
    #pragma unroll
    for (int j = 0; j < 4; ++j) {
      v16h vb = *(const v16h*)(&VlS[cur][j * 16 + m16][half * 16]);
      o[j] = wmma_f16(pa.v, vb, o[j]);
    }

    // async copies have had the whole compute section to land
    wait_async0();
    __syncthreads();
    cur ^= 1;
  }

  // --- normalize, repack to AO[(s*B + b)*E + h*64 + d] (f16) for out-proj
  const int b = bh / NHEADS;
  const int h = bh % NHEADS;
  #pragma unroll
  for (int j = 0; j < 4; ++j) {
    #pragma unroll
    for (int i = 0; i < 8; ++i) {
      const int srow = qbase + i + half * 8;
      const int r    = srow * BATCHV + b;
      const int col  = h * HDIM + j * 16 + m16;
      AO[(size_t)r * EMBED + col] = (_Float16)(o[j][i] / rsum[i]);
    }
  }
}

// ---------------------------------------------------------------------------
// Kernel 3: out = AO @ Wouth^T + bout (f32 result).  16(M) x 64(N) per wave.
// ---------------------------------------------------------------------------
__global__ void __launch_bounds__(32)
out_proj_kernel(const _Float16* __restrict__ AO, const _Float16* __restrict__ Wh,
                const float* __restrict__ bias, float* __restrict__ out)
{
  const int lane = threadIdx.x;
  const int half = lane >> 4;
  const int m16  = lane & 15;
  const int nbase = blockIdx.x * 64;   // over 512
  const int rbase = blockIdx.y * 16;   // over 8192

  v8f acc[4] = {};
  const _Float16* arow = AO + (size_t)(rbase + m16) * EMBED;

  for (int kb = 0; kb < EMBED; kb += 32) {
    AV a;
    a.h[0] = *(const v8h*)(arow + kb + half * 16);
    a.h[1] = *(const v8h*)(arow + kb + half * 16 + 8);
    v16h bt[4];
    #pragma unroll
    for (int j = 0; j < 4; ++j)
      bt[j] = *(const v16h*)(Wh + (size_t)(nbase + j * 16 + m16) * EMBED + kb + half * 16);
    #pragma unroll
    for (int j = 0; j < 4; ++j)
      acc[j] = wmma_f16(a.v, bt[j], acc[j]);
  }

  #pragma unroll
  for (int j = 0; j < 4; ++j) {
    const int n = nbase + j * 16 + m16;
    const float bv = bias[n];
    #pragma unroll
    for (int i = 0; i < 8; ++i)
      out[(size_t)(rbase + i + half * 8) * EMBED + n] = acc[j][i] + bv;
  }
}

// ---------------------------------------------------------------------------
extern "C" void kernel_launch(void* const* d_in, const int* in_sizes, int n_in,
                              void* d_out, int out_size, void* d_ws, size_t ws_size,
                              hipStream_t stream)
{
  const float* x    = (const float*)d_in[0];   // [S,B,E]
  const float* Win  = (const float*)d_in[1];   // [3E,E]
  const float* bin  = (const float*)d_in[2];   // [3E]
  const float* Wout = (const float*)d_in[3];   // [E,E]
  const float* bout = (const float*)d_in[4];   // [E]
  float* out = (float*)d_out;                  // [S,B,E]

  const size_t headBytes = (size_t)BHV * SEQV * HDIM * sizeof(_Float16); // 8 MB
  char* ws = (char*)d_ws;
  _Float16* Qh    = (_Float16*)(ws + 0 * headBytes);
  _Float16* Kh    = (_Float16*)(ws + 1 * headBytes);
  _Float16* Vt    = (_Float16*)(ws + 2 * headBytes);
  _Float16* AO    = (_Float16*)(ws + 3 * headBytes);
  _Float16* xhh   = (_Float16*)(ws + 4 * headBytes);                       // 8 MB
  _Float16* Winh  = (_Float16*)(ws + 5 * headBytes);                       // 1.5 MB
  _Float16* Wouth = (_Float16*)(ws + 5 * headBytes + (size_t)QKVN * EMBED * 2);

  const int nX  = ROWS * EMBED;        // 4194304
  const int nWi = QKVN * EMBED;        // 786432
  const int nWo = EMBED * EMBED;       // 262144
  cvt_f16_kernel<<<dim3((nX  + 255) / 256), dim3(256), 0, stream>>>(x,    xhh,   nX);
  cvt_f16_kernel<<<dim3((nWi + 255) / 256), dim3(256), 0, stream>>>(Win,  Winh,  nWi);
  cvt_f16_kernel<<<dim3((nWo + 255) / 256), dim3(256), 0, stream>>>(Wout, Wouth, nWo);

  const dim3 w32(32, 1, 1);
  qkv_pack_kernel <<<dim3(QKVN / 64, ROWS / 16, 1), w32, 0, stream>>>(xhh, Winh, bin, Qh, Kh, Vt);
  flash_attn_kernel<<<dim3(BHV, SEQV / 64, 1), dim3(128, 1, 1), 0, stream>>>(Qh, Kh, Vt, AO);
  out_proj_kernel <<<dim3(EMBED / 64, ROWS / 16, 1), w32, 0, stream>>>(AO, Wouth, bout, out);
}
